// VisualGraph_18622978196310
// MI455X (gfx1250) — compile-verified, matrix operand order checked by reading, not run on hardware
//
#include <hip/hip_runtime.h>
#include <hip/hip_bf16.h>

// MI455X / gfx1250, wave32. Heavy einsums run on v_wmma_f32_16x16x32_bf16.
// Staging: register-batched global_load_b128 -> ds_store_b128 with fixed trip
// counts (no exec juggling); next-chunk global_prefetch_b8 ahead of barriers.

typedef unsigned int u32;
typedef unsigned short u16;
typedef __attribute__((ext_vector_type(4)))  u32    u32x4;
typedef __attribute__((ext_vector_type(16))) __bf16 v16bf;
typedef __attribute__((ext_vector_type(8)))  __bf16 v8bf;
typedef __attribute__((ext_vector_type(8)))  float  v8f;

#define EPSV 1e-8f
#define NEGV -1e30f

// ---------- helpers ----------
__device__ __forceinline__ u16 f2bf(float x) {          // f32 -> bf16 (RNE)
  u32 u = __builtin_bit_cast(u32, x);
  u32 r = (u + 0x7FFFu + ((u >> 16) & 1u)) >> 16;
  return (u16)r;
}
__device__ __forceinline__ float bf2f(u16 s) {
  u32 u = ((u32)s) << 16;
  return __builtin_bit_cast(float, u);
}
// Build a 16x32 bf16 A/B fragment for this lane from two contiguous
// 16-byte LDS chunks (ISA 7.12.2: lanes<16 hold K{kb..kb+7,kb+16..kb+23}).
__device__ __forceinline__ v16bf ldfrag(const u16* p0, const u16* p1) {
  v8bf lo = *(const v8bf*)p0;
  v8bf hi = *(const v8bf*)p1;
  v16bf r;
#pragma unroll
  for (int i = 0; i < 8; i++) { r[i] = lo[i]; r[8 + i] = hi[i]; }
  return r;
}
__device__ __forceinline__ v8f wmma_bf16(v16bf a, v16bf b, v8f c) {
  return __builtin_amdgcn_wmma_f32_16x16x32_bf16(false, a, false, b,
                                                 (short)0, c, false, false);
}

// ---------- K0: weight-norm all MLP weights ----------
__global__ __launch_bounds__(256) void wnorm_kernel(
    const float* w1v, const float* w1g, const float* w2v, const float* w2g,
    const float* w9v, const float* w9g, const float* w10v, const float* w10g,
    float* w9n, float* w1n, float* w2n, float* w10n) {
  int t = threadIdx.x;
  __shared__ float nrm9[16], nrm1[8];
  if (t < 16) {
    float s = 0.f;
    for (int d = 0; d < 1024; d++) { float v = w9v[t * 1024 + d]; s += v * v; }
    nrm9[t] = w9g[t] / sqrtf(s);
  }
  if (t >= 32 && t < 40) {
    int o = t - 32; float s = 0.f;
    for (int k = 0; k < 32; k++) { float v = w1v[o * 32 + k]; s += v * v; }
    nrm1[o] = w1g[o] / sqrtf(s);
  }
  if (t == 64) {
    float s = 0.f;
    for (int k = 0; k < 8; k++) { float v = w2v[k]; s += v * v; }
    float sc = w2g[0] / sqrtf(s);
    for (int k = 0; k < 8; k++) w2n[k] = sc * w2v[k];
  }
  if (t == 96) {
    float s = 0.f;
    for (int k = 0; k < 16; k++) { float v = w10v[k]; s += v * v; }
    float sc = w10g[0] / sqrtf(s);
    for (int k = 0; k < 16; k++) w10n[k] = sc * w10v[k];
  }
  __syncthreads();
  for (int u = t; u < 16 * 1024; u += 256) w9n[u] = nrm9[u >> 10] * w9v[u];
  if (t < 256) { int u = t; w1n[u] = nrm1[u >> 5] * w1v[u]; }
}

// ---------- K1: caption softmax, tnodes bf16 (row-major + transposed copies),
//              rare_vector, text_few ----------
__global__ __launch_bounds__(256) void cap_kernel(
    const float* captions, const int* cap_lens, const float* word_freqs,
    const float* w9n, const float* b9, const float* w10n, const float* b10,
    u16* tnB, u16* tnBT, float* outTF, float* outRare) {
  int c = blockIdx.x, t = threadIdx.x, lane = t & 31, wv = t >> 5;
  __shared__ float sm[64];
  __shared__ float cap[1024];
  __shared__ float sR[16];
  int len = cap_lens[c];
  if (t == 0) {
    float mx = NEGV;
    for (int w = 0; w < len; w++) mx = fmaxf(mx, 0.02f * word_freqs[c * 50 + w]);
    float s = 0.f;
    for (int w = 0; w < len; w++) {
      float e = __expf(0.02f * word_freqs[c * 50 + w] - mx);
      sm[w] = e; s += e;
    }
    float inv = 1.f / s;
    for (int w = 0; w < len; w++) sm[w] *= inv;
    for (int w = len; w < 64; w++) sm[w] = 0.f;
  }
  __syncthreads();
  for (int w = 0; w < 50; w++) {
    for (int d = t; d < 1024; d += 256)
      cap[d] = captions[((size_t)c * 50 + w) * 1024 + d];
    __syncthreads();
    float scale = 1.0f + 0.02f * sm[w];
    for (int d = t; d < 1024; d += 256) {
      u16 b = f2bf(cap[d] * scale);
      tnB[((size_t)c * 64 + w) * 1024 + d] = b;                 // (C,64,1024)
      tnBT[((size_t)c * 1024 + d) * 64 + w] = b;                // (C,1024,64)
    }
    for (int o = wv; o < 16; o += 8) {                 // rare_vector row
      float acc = 0.f;
      for (int d = lane; d < 1024; d += 32) acc += cap[d] * w9n[o * 1024 + d];
#pragma unroll
      for (int m = 1; m < 32; m <<= 1) acc += __shfl_xor(acc, m, 32);
      if (lane == 0) sR[o] = acc + b9[o];
    }
    __syncthreads();
    if (t < 16) outRare[((size_t)c * 50 + w) * 16 + t] = sR[t];
    if (t == 0) {
      float s2 = b10[0];
      for (int o = 0; o < 16; o++) s2 += w10n[o] * tanhf(sR[o]);
      outTF[c * 50 + w] = s2;
    }
    __syncthreads();
  }
  for (int u = t; u < 14 * 1024; u += 256)             // zero pad rows 50..63
    tnB[((size_t)c * 64 + 50) * 1024 + u] = 0;
  for (int u = t; u < 1024 * 14; u += 256) {           // same for transposed copy
    int d = u / 14, w = 50 + (u % 14);
    tnBT[((size_t)c * 1024 + d) * 64 + w] = 0;
  }
}

// ---------- K2: images -> bf16, padded R 36->48 ----------
__global__ __launch_bounds__(256) void img_kernel(const float* images, u16* imB) {
  int i = blockIdx.x, t = threadIdx.x;
  for (int u = t; u < 36 * 1024; u += 256)
    imB[(size_t)i * 48 * 1024 + u] = f2bf(images[(size_t)i * 36 * 1024 + u]);
  for (int u = t; u < 12 * 1024; u += 256)
    imB[(size_t)i * 48 * 1024 + 36 * 1024 + u] = 0;
}

// ---------- K3: fused attention + wctx + block-cosine + MLP per (c,i) ----------
__global__ __launch_bounds__(128) void main_kernel(
    const u16* __restrict__ tnB, const u16* __restrict__ tnBT,
    const u16* __restrict__ imB,
    const int* __restrict__ cap_lens, const int* __restrict__ lam,
    const float* __restrict__ w1n, const float* __restrict__ b1,
    const float* __restrict__ w2n, const float* __restrict__ b2,
    float* __restrict__ outSim) {
  int c = blockIdx.x, i = blockIdx.y;
  int t = threadIdx.x, lane = t & 31, wv = t >> 5;
  float lambda_sm = (float)lam[0];

  __shared__ __align__(16) u16 sT[64][64];      // tnodes chunk (ph1 row-major, ph2 d-major)
  __shared__ __align__(16) u16 sI[48][64];      // images chunk
  __shared__ float sAttn[64][48];
  __shared__ __align__(16) u16 sAwB[48][64];    // softmaxed attention, bf16
  __shared__ float sNorm[64];
  __shared__ float sMv[48][32];
  __shared__ float sRed[64];

  const u16* tn = tnB + (size_t)c * 64 * 1024;
  const u16* im = imB + (size_t)i * 48 * 1024;
  const u32x4* gT4  = (const u32x4*)tn;                       // 128 u32x4 / row
  const u32x4* gI4  = (const u32x4*)im;
  const u32x4* gTT4 = (const u32x4*)(tnBT + (size_t)c * 1024 * 64); // 8 / row

  // ---- phase 1: attn[w][r] = sum_d tnodes[w][d]*images[r][d] (bf16 WMMA) ----
  v8f acc[3];
#pragma unroll
  for (int rt = 0; rt < 3; rt++)
#pragma unroll
    for (int j = 0; j < 8; j++) acc[rt][j] = 0.f;

  int kb8 = ((lane >> 4) << 3);                 // 0 or 8 half-wave K offset
  for (int kc = 0; kc < 16; kc++) {             // 16 chunks of 64 along D
    // register-batched staging: all b128 loads in flight, then LDS stores
    u32x4 rT[4], rI[3];
#pragma unroll
    for (int q = 0; q < 4; q++) {
      int u = t + 128 * q;
      rT[q] = gT4[(u >> 3) * 128 + kc * 8 + (u & 7)];
    }
#pragma unroll
    for (int q = 0; q < 3; q++) {
      int u = t + 128 * q;
      rI[q] = gI4[(u >> 3) * 128 + kc * 8 + (u & 7)];
    }
#pragma unroll
    for (int q = 0; q < 4; q++) ((u32x4*)sT)[t + 128 * q] = rT[q];
#pragma unroll
    for (int q = 0; q < 3; q++) ((u32x4*)sI)[t + 128 * q] = rI[q];
    if (kc < 15) {                              // prefetch next chunk into caches
      __builtin_prefetch((const void*)&gT4[(t >> 1) * 128 + (kc + 1) * 8], 0, 0);
      if (t < 96) __builtin_prefetch((const void*)&gI4[(t >> 1) * 128 + (kc + 1) * 8], 0, 0);
    }
    __syncthreads();
    int rowA = (wv << 4) + (lane & 15);
#pragma unroll
    for (int ks = 0; ks < 64; ks += 32) {
      v16bf a = ldfrag(&sT[rowA][ks + kb8], &sT[rowA][ks + kb8 + 16]);
#pragma unroll
      for (int rt = 0; rt < 3; rt++) {
        int rowB = (rt << 4) + (lane & 15);
        v16bf b = ldfrag(&sI[rowB][ks + kb8], &sI[rowB][ks + kb8 + 16]);
        acc[rt] = wmma_bf16(a, b, acc[rt]);
      }
    }
    __syncthreads();
  }
  // LeakyReLU(0.1) + scatter to LDS (C-layout: VGPR j -> M=j(+8 upper half))
  int mBase = (wv << 4) + ((lane >> 4) << 3);
#pragma unroll
  for (int rt = 0; rt < 3; rt++)
#pragma unroll
    for (int j = 0; j < 8; j++) {
      float v = acc[rt][j];
      sAttn[mBase + j][(rt << 4) + (lane & 15)] = (v >= 0.f) ? v : 0.1f * v;
    }
  __syncthreads();

  // ---- L2 norm over regions, masked softmax over words ----
  if (t < 64) {
    float s = 0.f;
    for (int r = 0; r < 36; r++) { float a = sAttn[t][r]; s += a * a; }
    sNorm[t] = 1.0f / (sqrtf(s) + EPSV);
  }
  __syncthreads();
  int len = cap_lens[c];
  if (t < 48) {
    float mx = NEGV;
    for (int w = 0; w < len; w++)
      mx = fmaxf(mx, lambda_sm * sAttn[w][t] * sNorm[w]);
    float sum = 0.f;
    for (int w = 0; w < len; w++)
      sum += __expf(lambda_sm * sAttn[w][t] * sNorm[w] - mx);
    float inv = 1.0f / sum;
    for (int w = 0; w < len; w++)
      sAwB[t][w] = f2bf(__expf(lambda_sm * sAttn[w][t] * sNorm[w] - mx) * inv);
    for (int w = len; w < 64; w++) sAwB[t][w] = 0;
  }
  __syncthreads();

  // ---- phase 2: wctx = aw @ tnodes, fused block-cosine into sMv ----
  for (int kc = 0; kc < 16; kc++) {
    u32x4 rT[4], rI[3];
#pragma unroll
    for (int q = 0; q < 4; q++)                 // transposed tnodes: linear src
      rT[q] = gTT4[kc * 512 + t + 128 * q];
#pragma unroll
    for (int q = 0; q < 3; q++) {
      int u = t + 128 * q;
      rI[q] = gI4[(u >> 3) * 128 + kc * 8 + (u & 7)];
    }
#pragma unroll
    for (int q = 0; q < 4; q++) ((u32x4*)sT)[t + 128 * q] = rT[q];
#pragma unroll
    for (int q = 0; q < 3; q++) ((u32x4*)sI)[t + 128 * q] = rI[q];
    if (kc < 15)
      __builtin_prefetch((const void*)&gTT4[(kc + 1) * 512 + t * 4], 0, 0);
    __syncthreads();
    if (wv < 3) {                               // wave-uniform: EXEC all-1s inside
      int rt = wv;
      int rowA = (rt << 4) + (lane & 15);
      for (int kbl = 0; kbl < 2; kbl++) {       // two 32-wide D blocks per chunk
        v8f a0, a1;
#pragma unroll
        for (int j = 0; j < 8; j++) { a0[j] = 0.f; a1[j] = 0.f; }
        int dcol = kbl * 32 + (lane & 15);
#pragma unroll
        for (int ks = 0; ks < 64; ks += 32) {   // contract over W=64
          v16bf a = ldfrag(&sAwB[rowA][ks + kb8], &sAwB[rowA][ks + kb8 + 16]);
          v16bf b0 = ldfrag(&sT[dcol][ks + kb8], &sT[dcol][ks + kb8 + 16]);
          a0 = wmma_bf16(a, b0, a0);
          v16bf b1v = ldfrag(&sT[dcol + 16][ks + kb8], &sT[dcol + 16][ks + kb8 + 16]);
          a1 = wmma_bf16(a, b1v, a1);
        }
        // block cosine: reduce wctx*img, wctx^2, img^2 over the 32-wide block
        int kbglob = kc * 2 + kbl;
        int mOff = (rt << 4) + ((lane >> 4) << 3);
#pragma unroll
        for (int j = 0; j < 8; j++) {
          int r = mOff + j;
          float w0 = a0[j], w1 = a1[j];
          float i0 = bf2f(sI[r][dcol]), i1 = bf2f(sI[r][dcol + 16]);
          float num = w0 * i0 + w1 * i1;
          float dw = w0 * w0 + w1 * w1;
          float dq = i0 * i0 + i1 * i1;
#pragma unroll
          for (int m = 1; m < 16; m <<= 1) {    // butterflies stay in 16-lane half
            num += __shfl_xor(num, m, 32);
            dw  += __shfl_xor(dw, m, 32);
            dq  += __shfl_xor(dq, m, 32);
          }
          if ((lane & 15) == 0)
            sMv[r][kbglob] = num / fmaxf(sqrtf(dw) * sqrtf(dq), EPSV);
        }
      }
    }
    __syncthreads();
  }

  // ---- tiny weight-normed MLP per region + mean over R ----
  if (t < 36) {
    float srow = b2[0];
    for (int o = 0; o < 8; o++) {
      float h = b1[o];
      for (int k = 0; k < 32; k++) h += w1n[o * 32 + k] * sMv[t][k];
      srow += w2n[o] * tanhf(h);
    }
    sRed[t] = srow;
  }
  __syncthreads();
  if (t == 0) {
    float s = 0.f;
    for (int r = 0; r < 36; r++) s += sRed[r];
    outSim[i * 64 + c] = s / 36.0f;             // similarities is (I,C)
  }
}

// ---------- launch ----------
extern "C" void kernel_launch(void* const* d_in, const int* in_sizes, int n_in,
                              void* d_out, int out_size, void* d_ws, size_t ws_size,
                              hipStream_t stream) {
  const float* images     = (const float*)d_in[0];
  const float* captions   = (const float*)d_in[1];
  // d_in[2] = bbox (unused by reference)
  const int*   cap_lens   = (const int*)d_in[3];
  const float* word_freqs = (const float*)d_in[4];
  // d_in[5]=embed_size, d_in[6]=num_block (shapes hardcoded)
  const int*   lam        = (const int*)d_in[7];
  const float* w1v  = (const float*)d_in[8];
  const float* w1g  = (const float*)d_in[9];
  const float* b1   = (const float*)d_in[10];
  const float* w2v  = (const float*)d_in[11];
  const float* w2g  = (const float*)d_in[12];
  const float* b2   = (const float*)d_in[13];
  const float* w9v  = (const float*)d_in[14];
  const float* w9g  = (const float*)d_in[15];
  const float* b9   = (const float*)d_in[16];
  const float* w10v = (const float*)d_in[17];
  const float* w10g = (const float*)d_in[18];
  const float* b10  = (const float*)d_in[19];

  float* out     = (float*)d_out;
  float* outSim  = out;                 // (64,64)
  float* outTF   = out + 4096;          // (64,50,1)
  float* outRare = out + 7296;          // (64,50,16)

  char* ws = (char*)d_ws;
  float* w9n  = (float*)(ws + 0);                  // 16*1024 f32
  float* w1n  = (float*)(ws + 65536);              // 8*32 f32
  float* w2n  = (float*)(ws + 66560);              // 8 f32
  float* w10n = (float*)(ws + 66592);              // 16 f32
  u16*   tnB  = (u16*)(ws + 131072);               // (64,64,1024) bf16
  u16*   imB  = (u16*)(ws + 131072 + 8388608);     // (64,48,1024) bf16
  u16*   tnBT = (u16*)(ws + 131072 + 8388608 + 6291456); // (64,1024,64) bf16

  wnorm_kernel<<<1, 256, 0, stream>>>(w1v, w1g, w2v, w2g, w9v, w9g, w10v, w10g,
                                      w9n, w1n, w2n, w10n);
  cap_kernel<<<64, 256, 0, stream>>>(captions, cap_lens, word_freqs,
                                     w9n, b9, w10n, b10, tnB, tnBT, outTF, outRare);
  img_kernel<<<64, 256, 0, stream>>>(images, imB);
  main_kernel<<<dim3(64, 64), 128, 0, stream>>>(tnB, tnBT, imB, cap_lens, lam,
                                                w1n, b1, w2n, b2, outSim);
}
